// AdditiveAttention_89996744720538
// MI455X (gfx1250) — matrix operand order repaired and checked by reading, not compiled, not run on hardware
//
#include <hip/hip_runtime.h>
#include <hip/hip_bf16.h>

// -------- types for CDNA5 WMMA --------
typedef __bf16 bf16_t;
typedef __attribute__((ext_vector_type(16))) __bf16 v16bf;
typedef __attribute__((ext_vector_type(8)))  float  v8f;

union Frag {
    int4  i[2];   // 2 x b128
    v16bf v;      // 16 bf16 = 32 bytes
};

#define NEG_INF_F (-1e9f)

// RNE float -> bf16
__device__ __forceinline__ unsigned short bf16_rne(float f) {
    union { float f; unsigned u; } x; x.f = f;
    unsigned u = x.u;
    unsigned r = u + 0x7FFFu + ((u >> 16) & 1u);
    return (unsigned short)(r >> 16);
}

// Branch-free tanh: prefer hardware V_TANH_F32 (CDNA5 TRANS32 op),
// else exp-based identity with no exec-mask divergence.
__device__ __forceinline__ float fast_tanh(float x) {
#if __has_builtin(__builtin_amdgcn_tanh_f32)
    return __builtin_amdgcn_tanh_f32(x);
#else
    float ax = __builtin_fabsf(x);
    float e  = __expf(-2.0f * ax);          // v_exp_f32
#if __has_builtin(__builtin_amdgcn_rcpf)
    float t  = (1.0f - e) * __builtin_amdgcn_rcpf(1.0f + e);
#else
    float t  = (1.0f - e) / (1.0f + e);
#endif
    return __builtin_copysignf(t, x);
#endif
}

// ---------------------------------------------------------------------------
// Kernel 1: proj_v[b][n] = sum_d vector[b][d] * W[d][n]   (32 x 1024, tiny)
// ---------------------------------------------------------------------------
__global__ void __launch_bounds__(256)
projv_kernel(const float* __restrict__ vec, const float* __restrict__ W,
             float* __restrict__ pv) {
    int idx = blockIdx.x * 256 + threadIdx.x;   // 0 .. 32767
    int b = idx >> 10;
    int n = idx & 1023;
    const float* vb = vec + (size_t)b * 1024;
    float acc = 0.f;
    #pragma unroll 8
    for (int d = 0; d < 1024; ++d)
        acc = fmaf(vb[d], W[(size_t)d * 1024 + n], acc);
    pv[idx] = acc;
}

// ---------------------------------------------------------------------------
// Kernel 2: Ut[n][k] = bf16(U[k][n])  — LDS-tiled transpose + convert.
// Makes WMMA B-fragments K-contiguous (two b128 loads per lane).
// ---------------------------------------------------------------------------
__global__ void __launch_bounds__(256)
u_transpose_bf16_kernel(const float* __restrict__ U, bf16_t* __restrict__ Ut) {
    __shared__ float tile[32][33];
    const int bx = blockIdx.x;          // n tile
    const int by = blockIdx.y;          // k tile
    const int tx = threadIdx.x;         // 0..31
    const int ty = threadIdx.y;         // 0..7
    #pragma unroll
    for (int i = 0; i < 4; ++i)
        tile[ty + i * 8][tx] = U[(size_t)(by * 32 + ty + i * 8) * 1024 + bx * 32 + tx];
    __syncthreads();
    #pragma unroll
    for (int i = 0; i < 4; ++i) {
        float f = tile[tx][ty + i * 8];
        unsigned short h = bf16_rne(f);
        union { unsigned short s; bf16_t b; } cv; cv.s = h;
        Ut[(size_t)(bx * 32 + ty + i * 8) * 1024 + by * 32 + tx] = cv.b;
    }
}

// ---------------------------------------------------------------------------
// Kernel 3: masked additive-attention logits via bf16 WMMA.
//   grid.x = B*R/32 = 4096 blocks, 256 threads = 8 wave32.
//   Workgroup: 32 matrix rows (read from HBM once -> bf16 LDS A-tile).
//   Wave nt (0..7): 16-wide N strip; computes BOTH 16-row M tiles against a
//   single B fragment (2 WMMAs per B load -> U L2 traffic halved).
//   N walked in 8 steps of 128; K = 1024 (32 B-loads, 64 WMMAs per strip).
// ---------------------------------------------------------------------------
__global__ void __launch_bounds__(256)
attn_scores_kernel(const float* __restrict__ matrix,
                   const int*   __restrict__ mask,
                   const bf16_t* __restrict__ Ut,
                   const float* __restrict__ pv,
                   const float* __restrict__ vvec,
                   float* __restrict__ out_logits) {
    __shared__ __align__(16) bf16_t Atile[32][1024];   // 64 KB
    __shared__ float part[8][32];

    const int tid  = threadIdx.x;
    const int row0 = blockIdx.x * 32;        // flat (b*R + r) row base
    const int b    = row0 >> 12;             // / 4096 (32 rows share b)

    // ---- Load A tile: 32 rows x 1024 f32 -> bf16 in LDS (HBM read once) ----
    {
        const float4* src = (const float4*)(matrix + (size_t)row0 * 1024);
        #pragma unroll 4
        for (int r = 0; r < 32; ++r) {
            float4 f = src[(size_t)r * 256 + tid];
            unsigned long long p =
                (unsigned long long)bf16_rne(f.x)
              | ((unsigned long long)bf16_rne(f.y) << 16)
              | ((unsigned long long)bf16_rne(f.z) << 32)
              | ((unsigned long long)bf16_rne(f.w) << 48);
            *(unsigned long long*)&Atile[r][tid * 4] = p;
        }
    }
    __syncthreads();

    const int lane  = tid & 31;
    const int nt    = tid >> 5;      // 0..7 : wave's 16-col strip in 128-step
    const int nl    = lane & 15;     // M for A-frag / N for B-frag & C
    const int khalf = lane >> 4;     // K-half selector (ISA wave32 layout)

    // A fragment bases (ISA: lane = khalf*16 + M; V0-3: K=khalf*8.., V4-7: +16)
    const bf16_t* abase0 = &Atile[nl][khalf * 8];        // M tile 0 (rows 0-15)
    const bf16_t* abase1 = &Atile[16 + nl][khalf * 8];   // M tile 1 (rows 16-31)

    float sacc0[8], sacc1[8];
    #pragma unroll
    for (int i = 0; i < 8; ++i) { sacc0[i] = 0.f; sacc1[i] = 0.f; }

    for (int nstep = 0; nstep < 8; ++nstep) {
        const int n = nstep * 128 + nt * 16 + nl;
        // B fragment base: column n of U == row n of Ut, K-contiguous.
        const bf16_t* bbase = Ut + (size_t)n * 1024 + khalf * 8;

        v8f acc0 = {};
        v8f acc1 = {};
        #pragma unroll 4
        for (int k0 = 0; k0 < 1024; k0 += 32) {
            Frag A0, A1, B;
            B.i[0]  = *(const int4*)(bbase + k0);        // K = k0+khalf*8 .. +7
            B.i[1]  = *(const int4*)(bbase + k0 + 16);   // K = k0+16+khalf*8 ..
            A0.i[0] = *(const int4*)(abase0 + k0);       // K = k0+khalf*8 .. +7
            A0.i[1] = *(const int4*)(abase0 + k0 + 16);  // K = k0+16+khalf*8 ..
            A1.i[0] = *(const int4*)(abase1 + k0);
            A1.i[1] = *(const int4*)(abase1 + k0 + 16);
            acc0 = __builtin_amdgcn_wmma_f32_16x16x32_bf16(
                       false, A0.v, false, B.v, (short)0, acc0, false, false);
            acc1 = __builtin_amdgcn_wmma_f32_16x16x32_bf16(
                       false, A1.v, false, B.v, (short)0, acc1, false, false);
        }
        // Epilogue: lane holds C[m][n] for m = 8*khalf + i (per M tile), fixed n.
        const float pvn = pv[(size_t)b * 1024 + n];
        const float vn  = vvec[n];
        #pragma unroll
        for (int i = 0; i < 8; ++i) {
            sacc0[i] += fast_tanh(pvn + acc0[i]) * vn;
            sacc1[i] += fast_tanh(pvn + acc1[i]) * vn;
        }
    }

    // Reduce over the 16 N-lanes of each khalf group (wave32: masks 1,2,4,8).
    #pragma unroll
    for (int m = 1; m <= 8; m <<= 1)
        #pragma unroll
        for (int i = 0; i < 8; ++i) {
            sacc0[i] += __shfl_xor(sacc0[i], m, 32);
            sacc1[i] += __shfl_xor(sacc1[i], m, 32);
        }

    if (nl == 0) {
        #pragma unroll
        for (int i = 0; i < 8; ++i) {
            part[nt][khalf * 8 + i]      = sacc0[i];   // rows 0-15
            part[nt][16 + khalf * 8 + i] = sacc1[i];   // rows 16-31
        }
    }
    __syncthreads();

    if (tid < 32) {
        float s = 0.f;
        #pragma unroll
        for (int w = 0; w < 8; ++w) s += part[w][tid];
        int row = row0 + tid;                 // flat b*4096 + r
        out_logits[row] = (mask[row] > 0) ? s : NEG_INF_F;
    }
}

// ---------------------------------------------------------------------------
// Kernel 4: in-place masked softmax over R=4096, one block per batch row.
// ---------------------------------------------------------------------------
__global__ void __launch_bounds__(256)
softmax_kernel(float* __restrict__ out) {
    __shared__ float red[256];
    const int b   = blockIdx.x;
    const int tid = threadIdx.x;
    float* row = out + (size_t)b * 4096;

    float x[16];
    float lmax = -__builtin_inff();
    #pragma unroll
    for (int j = 0; j < 16; ++j) {
        x[j] = row[tid + j * 256];
        lmax = fmaxf(lmax, x[j]);
    }
    red[tid] = lmax;
    __syncthreads();
    for (int s = 128; s > 0; s >>= 1) {
        if (tid < s) red[tid] = fmaxf(red[tid], red[tid + s]);
        __syncthreads();
    }
    const float m = red[0];
    __syncthreads();

    float lsum = 0.f;
    #pragma unroll
    for (int j = 0; j < 16; ++j) {
        x[j] = __expf(x[j] - m);
        lsum += x[j];
    }
    red[tid] = lsum;
    __syncthreads();
    for (int s = 128; s > 0; s >>= 1) {
        if (tid < s) red[tid] += red[tid + s];
        __syncthreads();
    }
    const float inv = 1.0f / red[0];
    #pragma unroll
    for (int j = 0; j < 16; ++j)
        row[tid + j * 256] = x[j] * inv;
}

// ---------------------------------------------------------------------------
extern "C" void kernel_launch(void* const* d_in, const int* in_sizes, int n_in,
                              void* d_out, int out_size, void* d_ws, size_t ws_size,
                              hipStream_t stream) {
    (void)in_sizes; (void)n_in; (void)out_size; (void)ws_size;

    const float* vec    = (const float*)d_in[0];   // (32, 1024)
    const float* matrix = (const float*)d_in[1];   // (32, 4096, 1024)
    const int*   mmask  = (const int*)  d_in[2];   // (32, 4096)
    const float* W      = (const float*)d_in[3];   // (1024, 1024)
    const float* U      = (const float*)d_in[4];   // (1024, 1024)
    const float* vv     = (const float*)d_in[5];   // (1024, 1)
    float* out = (float*)d_out;                    // (32, 4096)

    // Workspace layout: [ Ut bf16 : 2 MB ][ proj_v f32 : 128 KB ]
    bf16_t* Ut = (bf16_t*)d_ws;
    float*  pv = (float*)((char*)d_ws + (size_t)1024 * 1024 * 2);

    projv_kernel<<<128, 256, 0, stream>>>(vec, W, pv);
    u_transpose_bf16_kernel<<<dim3(32, 32), dim3(32, 8), 0, stream>>>(U, Ut);
    attn_scores_kernel<<<4096, 256, 0, stream>>>(matrix, mmask, Ut, pv, vv, out);
    softmax_kernel<<<32, 256, 0, stream>>>(out);
}
